// CFP_7095285973755
// MI455X (gfx1250) — matrix operand order repaired and checked by reading, not compiled, not run on hardware
//
#include <hip/hip_runtime.h>
#include <cstdint>

// MI455X / gfx1250, wave32. Flash-attention formulation of:
//   x = Linear(inp^T); S = (x*scale) @ x^T; attn = softmax(S); out = attn @ flow
// Matmuls via v_wmma_f32_16x16x32_f16 (fp32 accumulate); key tiles staged to
// LDS by the Tensor Data Mover (tensor_load_to_lds + TENSORcnt), double-buffered.

typedef _Float16 v16h __attribute__((ext_vector_type(16)));
typedef _Float16 v8h  __attribute__((ext_vector_type(8)));
typedef float    v8f  __attribute__((ext_vector_type(8)));
typedef unsigned int v4u __attribute__((ext_vector_type(4)));
typedef int          v4i __attribute__((ext_vector_type(4)));
typedef int          v8i __attribute__((ext_vector_type(8)));

#define NB   4
#define CC   256
#define NN   4096          // H*W
#define L2E  1.44269504088896340736f
#define KSTR 272           // LDS row stride in halves: 544 B = 34*16 (16B-aligned rows)
#define KTILE_HALVES (64 * KSTR)           // one 64-row key tile in LDS
#define KTILE_BYTES  (KTILE_HALVES * 2)    // 34816 B

#if __has_builtin(__builtin_amdgcn_tensor_load_to_lds) && \
    __has_builtin(__builtin_amdgcn_s_wait_tensorcnt)
#define HAVE_TDM 1
#else
#define HAVE_TDM 0
#endif

// Build a v16h from two aligned 16-byte chunks (2x global/ds _load_b128).
__device__ __forceinline__ v16h load16h(const _Float16* __restrict__ p0,
                                        const _Float16* __restrict__ p1) {
  v8h lo = *(const v8h*)p0;
  v8h hi = *(const v8h*)p1;
  v16h r;
#pragma unroll
  for (int i = 0; i < 8; ++i) { r[i] = lo[i]; r[i + 8] = hi[i]; }
  return r;
}

#if HAVE_TDM
// TDM D# for a 2D tile: 64 rows x 256 halves (512 B) from Xs[b] (row stride
// 256 halves) into LDS at lds_byte_addr, padding +8 dwords every 128 dwords
// so the LDS row stride becomes 272 halves == KSTR.
__device__ __forceinline__ void tdm_load_ktile(unsigned lds_byte_addr,
                                               const _Float16* gsrc) {
  const unsigned long long ga = (unsigned long long)(uintptr_t)gsrc;
  v4u g0;
  g0[0] = 1u;                                   // count=1, user descriptor
  g0[1] = lds_byte_addr;                        // lds_addr
  g0[2] = (unsigned)(ga & 0xffffffffu);         // global_addr[31:0]
  g0[3] = (unsigned)((ga >> 32) & 0x01ffffffu)  // global_addr[56:32]
          | (2u << 30);                         // type = 2 ("image")
  v8i g1;
  g1[0] = (1 << 16)      // data_size = 1 -> 2 bytes
        | (1 << 20)      // pad_enable
        | (6 << 22)      // pad_interval: 2^(6+1) = 128 dwords (= one 512B row)
        | (7 << 25);     // pad_amount: 7+1 = 8 dwords (= 32 B -> KSTR rows)
  g1[1] = (int)(256u << 16);   // tensor_dim0 = 256 (lo16)
  g1[2] = (int)(4096u << 16);  // tensor_dim0 hi = 0 | tensor_dim1 = 4096 (lo16)
  g1[3] = (int)(256u << 16);   // tensor_dim1 hi = 0 | tile_dim0 = 256
  g1[4] = 64;                  // tile_dim1 = 64, tile_dim2 = 0
  g1[5] = 256;                 // tensor_dim0_stride[31:0] = 256 elements
  g1[6] = 0;                   // stride hi / tensor_dim1_stride lo (unused, 2D)
  g1[7] = 0;
  const v4i z4 = {0, 0, 0, 0};
#if defined(__clang_major__) && __clang_major__ >= 23
  const v8i z8 = {0, 0, 0, 0, 0, 0, 0, 0};
  __builtin_amdgcn_tensor_load_to_lds(g0, g1, z4, z4, z8, 0);
#else
  __builtin_amdgcn_tensor_load_to_lds(g0, g1, z4, z4, 0);
#endif
}
#endif

// ---------------------------------------------------------------------------
// Kernel 1: Xs[b, n, c] = ((W @ inp_b)[c, n] + bias[c]) * 0.25  (f16)
// One wave = one 16(c) x 16(n) tile of X^T = W @ inp + b. inp slab staged to
// LDS transposed; A-tiles straight from W rows (f32 -> f16 inline).
// ---------------------------------------------------------------------------
__global__ __launch_bounds__(128) void linear_kernel(
    const float* __restrict__ inp,   // [B, C, N]
    const float* __restrict__ Wl,    // [C, C] (out, in) row-major
    const float* __restrict__ bl,    // [C]
    _Float16* __restrict__ Xs)       // [B, N, C] f16, pre-scaled by 1/4
{
  __shared__ _Float16 sX[16 * KSTR];          // [n][k], padded rows

  const int t     = blockIdx.x;               // 0 .. B*1024-1
  const int b     = t >> 10;
  const int rem   = t & 1023;
  const int ntile = rem & 255;                // 16-row slab of N
  const int cgrp  = rem >> 8;                 // 64-channel group
  const int tid   = threadIdx.x;
  const int lane  = tid & 31;
  const int wave  = tid >> 5;
  const int nl    = lane & 15;
  const int hi16  = (lane >> 4) & 1;

  const float* src = inp + (size_t)b * CC * NN + ntile * 16;
#pragma unroll
  for (int rep = 0; rep < 32; ++rep) {
    int e = rep * 128 + tid;                  // 0..4095
    int k = e >> 4;
    int n = e & 15;
    sX[n * KSTR + k] = (_Float16)src[(size_t)k * NN + n];
  }
  __syncthreads();

  const int ctile = cgrp * 64 + wave * 16;
  const int crow  = ctile + nl;               // A-matrix row (c)
  const int aoff  = hi16 ? 8 : 0;
  const int boff  = hi16 ? 8 : 0;

  v8f acc = {};
#pragma unroll
  for (int kk = 0; kk < 8; ++kk) {
    const int k0 = kk * 32;
    const float* wr = Wl + (size_t)crow * CC + k0 + aoff;
    v16h a;
#pragma unroll
    for (int i = 0; i < 8; ++i) {
      a[i]     = (_Float16)wr[i];
      a[i + 8] = (_Float16)wr[16 + i];
    }
    const _Float16* bp = &sX[nl * KSTR + k0 + boff];
    v16h bt = load16h(bp, bp + 16);
    acc = __builtin_amdgcn_wmma_f32_16x16x32_f16(false, a, false, bt,
                                                 (short)0, acc, false, false);
  }

  const int nglob = ntile * 16 + nl;
  _Float16* xrow = Xs + ((size_t)b * NN + nglob) * CC;
  const int chalf = hi16 ? 8 : 0;
#pragma unroll
  for (int r = 0; r < 8; ++r) {
    const int c = ctile + r + chalf;
    xrow[c] = (_Float16)((acc[r] + bl[c]) * 0.25f);   // fold scale (1/4)^2 = 1/16
  }
}

// ---------------------------------------------------------------------------
// Kernel 2: streaming attention. Block = 4 waves (same batch), each wave owns
// 16 query rows (Q in registers). Per iteration: one 64-key-column tile is
// DMA'd into LDS by the TDM (double-buffered, overlapped with compute), then
// 4 x 8 WMMAs + one online-softmax update with shfl_xor reductions.
// ---------------------------------------------------------------------------
__global__ __launch_bounds__(128) void attn_flow_kernel(
    const _Float16* __restrict__ Xs,   // [B, N, C] f16, scaled by 1/4
    const float* __restrict__ flow,    // [B, 2, N]
    float* __restrict__ out)           // [B, 2, N]
{
  __shared__ _Float16 sK[2 * KTILE_HALVES];   // double-buffered key tiles

  const int tid  = threadIdx.x;
  const int lane = tid & 31;
  const int wave = tid >> 5;
  const int g    = blockIdx.x * 4 + wave;     // 0..1023 row-blocks
  const int b    = g >> 8;                    // uniform across the block
  const int rb   = g & 255;
  const int nl   = lane & 15;
  const int hi16 = (lane >> 4) & 1;

  const _Float16* xb = Xs + (size_t)b * NN * CC;

#if HAVE_TDM
  if (wave == 0)                               // kick off tile 0 early
    tdm_load_ktile((unsigned)(uintptr_t)&sK[0], xb);
#endif

  // Q: A-layout. lane nl = query row; elems 0..7 at K+(hi16?8:0), 8..15 at +16.
  const _Float16* qrow = xb + (size_t)(rb * 16 + nl) * CC;
  v16h q[8];
#pragma unroll
  for (int kk = 0; kk < 8; ++kk) {
    const _Float16* p = qrow + kk * 32 + hi16 * 8;
    q[kk] = load16h(p, p + 16);
  }

  const float* fxp = flow + (size_t)b * 2 * NN;
  const float* fyp = fxp + NN;

  float mrow[8], lsum[8], axv[8], ayv[8];
#pragma unroll
  for (int r = 0; r < 8; ++r) {
    mrow[r] = -__builtin_inff(); lsum[r] = 0.f; axv[r] = 0.f; ayv[r] = 0.f;
  }

  for (int ct = 0; ct < NN / 64; ++ct) {
    const int cur = ct & 1;
#if HAVE_TDM
    if (wave == 0)
      __builtin_amdgcn_s_wait_tensorcnt(0);    // tile ct resident in LDS
    __syncthreads();                           // everyone sees it; prev reads done
    if (wave == 0 && ct + 1 < NN / 64)         // overlap DMA of ct+1 with compute
      tdm_load_ktile((unsigned)(uintptr_t)&sK[((ct + 1) & 1) * KTILE_HALVES],
                     xb + (size_t)(ct + 1) * 64 * CC);
#else
    __syncthreads();                           // previous readers of this buffer done
    {                                          // cooperative staging fallback
      const _Float16* srcp = xb + (size_t)ct * 64 * CC;
#pragma unroll
      for (int rep = 0; rep < 16; ++rep) {
        int chunk = rep * 128 + tid;           // 2048 x 16B chunks
        int row = chunk >> 5;
        int co  = (chunk & 31) * 8;
        *(v8h*)&sK[cur * KTILE_HALVES + row * KSTR + co] =
            *(const v8h*)&srcp[(size_t)row * CC + co];
      }
    }
    __syncthreads();
#endif

    const _Float16* kbase = &sK[cur * KTILE_HALVES];
    const int col0 = ct * 64;
    v8f s[4];
    float fx[4], fy[4];
#pragma unroll
    for (int sb = 0; sb < 4; ++sb) {
      fx[sb] = fxp[col0 + sb * 16 + nl];
      fy[sb] = fyp[col0 + sb * 16 + nl];
      // B tile = K^T from LDS: lane nl = key row; elems 0..7 at 32kk + hi16*8, 8..15 at +16
      const _Float16* krow = kbase + (sb * 16 + nl) * KSTR + hi16 * 8;
      v8f acc = {};
#pragma unroll
      for (int kk = 0; kk < 8; ++kk) {
        const _Float16* p = krow + kk * 32;
        v16h kt = load16h(p, p + 16);
        acc = __builtin_amdgcn_wmma_f32_16x16x32_f16(false, q[kk], false, kt,
                                                     (short)0, acc, false, false);
      }
      s[sb] = acc;
    }

    // Online softmax over these 64 columns (rows r / r+8 per lane half).
#pragma unroll
    for (int r = 0; r < 8; ++r) {
      float tmax = fmaxf(fmaxf(s[0][r], s[1][r]), fmaxf(s[2][r], s[3][r]));
#pragma unroll
      for (int off = 8; off >= 1; off >>= 1)
        tmax = fmaxf(tmax, __shfl_xor(tmax, off, 32));
      const float mnew = fmaxf(mrow[r], tmax);
      const float corr = __builtin_amdgcn_exp2f((mrow[r] - mnew) * L2E);
      mrow[r] = mnew;
      const float p0 = __builtin_amdgcn_exp2f((s[0][r] - mnew) * L2E);
      const float p1 = __builtin_amdgcn_exp2f((s[1][r] - mnew) * L2E);
      const float p2 = __builtin_amdgcn_exp2f((s[2][r] - mnew) * L2E);
      const float p3 = __builtin_amdgcn_exp2f((s[3][r] - mnew) * L2E);
      float ps = p0 + p1 + p2 + p3;
      float px = p0 * fx[0] + p1 * fx[1] + p2 * fx[2] + p3 * fx[3];
      float py = p0 * fy[0] + p1 * fy[1] + p2 * fy[2] + p3 * fy[3];
#pragma unroll
      for (int off = 8; off >= 1; off >>= 1) {
        ps += __shfl_xor(ps, off, 32);
        px += __shfl_xor(px, off, 32);
        py += __shfl_xor(py, off, 32);
      }
      lsum[r] = lsum[r] * corr + ps;
      axv[r]  = axv[r]  * corr + px;
      ayv[r]  = ayv[r]  * corr + py;
    }
  }

  // Lane 0 holds rows rb*16+0..7, lane 16 holds rows rb*16+8..15 (replicated).
  if (nl == 0) {
    float* ox = out + (size_t)b * 2 * NN + rb * 16 + hi16 * 8;
    float* oy = ox + NN;
#pragma unroll
    for (int r = 0; r < 8; ++r) {
      const float inv = 1.0f / lsum[r];
      ox[r] = axv[r] * inv;
      oy[r] = ayv[r] * inv;
    }
  }
}

// ---------------------------------------------------------------------------
extern "C" void kernel_launch(void* const* d_in, const int* in_sizes, int n_in,
                              void* d_out, int out_size, void* d_ws, size_t ws_size,
                              hipStream_t stream) {
  (void)in_sizes; (void)n_in; (void)out_size; (void)ws_size;
  const float* inp  = (const float*)d_in[0];
  // d_in[1] = inter_mask: unused by the reference computation.
  const float* flow = (const float*)d_in[2];
  const float* Wl   = (const float*)d_in[3];
  const float* bl   = (const float*)d_in[4];
  float* out = (float*)d_out;
  _Float16* Xs = (_Float16*)d_ws;            // B*N*C f16 = 8 MiB scratch

  linear_kernel<<<NB * 1024, 128, 0, stream>>>(inp, Wl, bl, Xs);
  attn_flow_kernel<<<256, 128, 0, stream>>>(Xs, flow, out);
}
